// SSVA_78838419685485
// MI455X (gfx1250) — matrix operand order repaired and checked by reading, not compiled, not run on hardware
//
#include <hip/hip_runtime.h>
#include <hip/hip_bf16.h>

typedef __attribute__((ext_vector_type(8)))  _Float16 half8;
typedef __attribute__((ext_vector_type(16))) _Float16 v16h;
typedef __attribute__((ext_vector_type(8)))  float    v8f;

#define NTOK 16384            // B*H*W
#define NT   163840           // NTOK * T
#define SZ   10485760ull      // B*H*W*D*T elements

// ---------- WMMA helpers (CDNA5 gfx1250, wave32) ----------
// A fragment 16x32 f16: lane l holds M=l&15; element e -> K = (l>>4)*16 + e
// => two contiguous 8-half groups at base +0 and +8 (base includes (l>>4)*16).
// B fragment 32x16 f16: lane l holds N=l&15; element e -> K = (l>>4)*16 + e => 16 contiguous halves.
// C/D 16x16 f32: lane l holds N=l&15; VGPR v -> M = (l>>4)*8 + v.
__device__ __forceinline__ v16h mk_afrag(half8 lo, half8 hi) {
  v16h r;
#pragma unroll
  for (int i = 0; i < 8; ++i) { r[i] = lo[i]; r[i + 8] = hi[i]; }
  return r;
}

__device__ __forceinline__ v8f wmma16(v16h a, v16h b, v8f c) {
  return __builtin_amdgcn_wmma_f32_16x16x32_f16(false, a, false, b, (short)0, c, false, false);
}

// ---------- BN folding: a = gamma*rsqrt(var+eps); b = (convbias-mean)*a + beta ----------
struct ScaleArgs {
  const float* g[7]; const float* be[7]; const float* mn[7]; const float* vr[7]; const float* bias[7];
  int cnt[7]; int off[7];
  float* out;
};

__global__ void prep_scales_k(ScaleArgs A) {
  int tid = blockIdx.x * blockDim.x + threadIdx.x;
  int set = -1, ch = 0, acc = 0;
#pragma unroll
  for (int s = 0; s < 7; ++s) {
    if (set < 0 && tid < acc + A.cnt[s]) { set = s; ch = tid - acc; }
    acc += A.cnt[s];
  }
  if (set < 0) return;
  float a = A.g[set][ch] * rsqrtf(A.vr[set][ch] + 1e-5f);
  float bia = A.bias[set] ? A.bias[set][ch] : 0.f;
  float b = (bia - A.mn[set][ch]) * a + A.be[set][ch];
  A.out[A.off[set] + ch] = a;
  A.out[A.off[set] + A.cnt[set] + ch] = b;
}

// ---------- weights -> f16 [Nout][K] layouts ----------
// wSpdc [64][1600] k=(dy*5+dx)*64+c @0 ; wTemp [64][320] k=dt*64+c @102400 ;
// wConv @122880 ; wProj @126976 ; wF1 [128][64] @131072 ; wF2 [64][128] @139264 ; total 147456
__global__ void prep_weights_k(const float* spdc_w, const float* temp_w, const float* conv_w,
                               const float* proj_w, const float* f1_w, const float* f2_w,
                               _Float16* wb) {
  int i = blockIdx.x * blockDim.x + threadIdx.x;
  if (i < 102400) {
    int dout = i / 1600, k = i % 1600;
    int dy = k / 320, dx = (k / 64) % 5, c = k % 64;
    wb[i] = (_Float16)spdc_w[((dout * 64 + c) * 5 + dy) * 5 + dx];
  } else if (i < 122880) {
    int j = i - 102400;
    int dout = j / 320, k = j % 320, dt = k / 64, c = k % 64;
    wb[i] = (_Float16)temp_w[(dout * 64 + c) * 5 + dt];
  } else if (i < 126976) wb[i] = (_Float16)conv_w[i - 122880];
  else if (i < 131072)   wb[i] = (_Float16)proj_w[i - 126976];
  else if (i < 139264)   wb[i] = (_Float16)f1_w[i - 131072];
  else if (i < 147456)   wb[i] = (_Float16)f2_w[i - 139264];
}

// ---------- x -> xhwc f16 [B][T][H][W][D] and V f32 [N][T][D] ----------
__global__ void prep_xv_k(const float* x, _Float16* xhwc, float* vbuf) {
  int tid = blockIdx.x * blockDim.x + threadIdx.x;
  int tok = tid >> 6, d = tid & 63;
  if (tok >= NTOK) return;
  int b = tok >> 12, hw = tok & 4095;
  const float* xp = x + ((size_t)tok * 64 + d) * 10;
  float xv[10];
#pragma unroll
  for (int t = 0; t < 10; ++t) xv[t] = xp[t];
#pragma unroll
  for (int t = 0; t < 10; ++t) {
    xhwc[((size_t)(b * 10 + t) * 4096 + hw) * 64 + d] = (_Float16)xv[t];
    vbuf[(size_t)tok * 640 + t * 64 + d] = xv[t];
  }
}

// ---------- spdc 5x5 spatial conv: implicit GEMM, K=1600 (50 chunks) ----------
// Block handles 2 image rows (h0, h0+1); each wave: M=32 pixels x N=64 outputs,
// so each B fragment is reused by two WMMAs (12 memory ops : 8 WMMAs per chunk).
__global__ void __launch_bounds__(128) spdc_k(const _Float16* xhwc, const _Float16* wspdc,
                                              const float* sc, _Float16* q1h) {
  __shared__ _Float16 strip[6 * 68 * 64];   // rows h0-2..h0+3, cols -2..65, zero padded
  int bid = blockIdx.x;
  int b = bid / 320, rem = bid % 320, t = rem / 32, hb = rem % 32;
  int h0 = hb * 2;
  int tid = threadIdx.x;
  for (int p = tid; p < 408; p += 128) {
    int dy = p / 68, col = p % 68;
    int hr = h0 + dy - 2, wr = col - 2;
    uint4* dst = (uint4*)&strip[p * 64];
    if (hr >= 0 && hr < 64 && wr >= 0 && wr < 64) {
      const uint4* src = (const uint4*)(xhwc + ((size_t)(b * 10 + t) * 64 + hr) * 4096 + (size_t)wr * 64);
#pragma unroll
      for (int i = 0; i < 8; ++i) dst[i] = src[i];
    } else {
      uint4 z = make_uint4(0, 0, 0, 0);
#pragma unroll
      for (int i = 0; i < 8; ++i) dst[i] = z;
    }
  }
  __syncthreads();
  int lane = tid & 31, wave = tid >> 5, khalf = lane >> 4;
  int ry = wave >> 1;            // which of the two output rows
  int m0 = (wave & 1) * 32;      // 32-pixel column group
  v8f acc[2][4] = {};
  for (int kc = 0; kc < 50; ++kc) {
    int dy = kc / 10, rm = kc % 10, dx = rm >> 1, c0 = (rm & 1) * 32;
    int ab = ((ry + dy) * 68 + m0 + (lane & 15) + dx) * 64 + c0 + khalf * 16;
    v16h a0 = mk_afrag(*(const half8*)&strip[ab], *(const half8*)&strip[ab + 8]);
    v16h a1 = mk_afrag(*(const half8*)&strip[ab + 1024], *(const half8*)&strip[ab + 1032]);
#pragma unroll
    for (int nt = 0; nt < 4; ++nt) {
      const _Float16* bp = wspdc + (nt * 16 + (lane & 15)) * 1600 + kc * 32 + khalf * 16;
      v16h bf = *(const v16h*)bp;
      acc[0][nt] = wmma16(a0, bf, acc[0][nt]);
      acc[1][nt] = wmma16(a1, bf, acc[1][nt]);
    }
  }
  const float* aS = sc;  const float* bS = sc + 64;
  int h = h0 + ry;
#pragma unroll
  for (int f = 0; f < 2; ++f) {
#pragma unroll
    for (int nt = 0; nt < 4; ++nt) {
      int d = nt * 16 + (lane & 15);
      float av = aS[d], bv = bS[d];
#pragma unroll
      for (int v = 0; v < 8; ++v) {
        int wv = m0 + f * 16 + khalf * 8 + v;
        float val = av * acc[f][nt][v] + bv;
        val = val > 0.f ? val : 0.f;
        q1h[(((size_t)(b * 10 + t) * 64 + h) * 64 + wv) * 64 + d] = (_Float16)val;
      }
    }
  }
}

// ---------- temporal 5-tap conv: implicit GEMM, K=320 (10 chunks, zero-pad in t) ----------
__global__ void __launch_bounds__(128) temp_k(const _Float16* q1h, const _Float16* wtemp,
                                              const float* sc, float* qbuf) {
  int bid = blockIdx.x;
  int b = bid / 640, rem = bid % 640, t = rem / 64, h = rem % 64;
  int tid = threadIdx.x, lane = tid & 31, wave = tid >> 5, khalf = lane >> 4;
  int wp = wave * 16 + (lane & 15);
  v8f acc[4] = {};
  for (int kc = 0; kc < 10; ++kc) {
    int dt = kc >> 1, c0 = (kc & 1) * 32;
    int tt = t + dt - 2;
    if (tt < 0 || tt >= 10) continue;
    const _Float16* ap = q1h + (((size_t)(b * 10 + tt) * 64 + h) * 64 + wp) * 64 + c0 + khalf * 16;
    v16h a = mk_afrag(*(const half8*)ap, *(const half8*)(ap + 8));
#pragma unroll
    for (int nt = 0; nt < 4; ++nt) {
      const _Float16* bp = wtemp + (nt * 16 + (lane & 15)) * 320 + kc * 32 + khalf * 16;
      acc[nt] = wmma16(a, *(const v16h*)bp, acc[nt]);
    }
  }
  const float* aT = sc + 128; const float* bT = sc + 192;
#pragma unroll
  for (int nt = 0; nt < 4; ++nt) {
    int d = nt * 16 + (lane & 15);
    float av = aT[d], bv = bT[d];
#pragma unroll
    for (int v = 0; v < 8; ++v) {
      int wv = wave * 16 + khalf * 8 + v;
      float val = av * acc[nt][v] + bv;
      qbuf[((size_t)(b * 10 + t) * 4096 + h * 64 + wv) * 64 + d] = val > 0.f ? val : 0.f;
    }
  }
}

// ---------- learnable WT: per-head 10x10 temporal transform + bn1 + relu (VALU) ----------
__global__ void wt1_k(const float* x, const float* WT, const float* sc, _Float16* ybuf) {
  __shared__ float sWT[800];
  for (int i = threadIdx.x; i < 800; i += 256) sWT[i] = WT[i];
  __syncthreads();
  int tok = blockIdx.x * 4 + (threadIdx.x >> 6);
  int d = threadIdx.x & 63, head = d >> 3;
  const float* xp = x + ((size_t)tok * 64 + d) * 10;
  float xv[10];
#pragma unroll
  for (int j = 0; j < 10; ++j) xv[j] = xp[j];
  float a = sc[256 + d], bb = sc[320 + d];
#pragma unroll
  for (int i = 0; i < 10; ++i) {
    float s = 0.f;
#pragma unroll
    for (int j = 0; j < 10; ++j) s += sWT[(head * 10 + i) * 10 + j] * xv[j];
    float val = a * s + bb;
    val = val > 0.f ? val : 0.f;
    ybuf[(size_t)tok * 640 + i * 64 + d] = (_Float16)val;
  }
}

// ---------- WT 1x1 conv + bn2 + relu -> K in natural [B,D,T,H,W] (scrambled) layout ----------
__global__ void __launch_bounds__(128) wt2_k(const _Float16* ybuf, const _Float16* wconv,
                                             const float* sc, float* kbuf) {
  __shared__ float tile[64 * 66];
  int bid = blockIdx.x;
  int b = bid / 640, rem = bid % 640, h = rem / 10, t = rem % 10;
  int tid = threadIdx.x, lane = tid & 31, wave = tid >> 5, khalf = lane >> 4;
  int wp = wave * 16 + (lane & 15);
  v8f acc[4] = {};
#pragma unroll
  for (int kc = 0; kc < 2; ++kc) {
    const _Float16* ap = ybuf + (((size_t)(b * 64 + h) * 64 + wp) * 10 + t) * 64 + kc * 32 + khalf * 16;
    v16h a = mk_afrag(*(const half8*)ap, *(const half8*)(ap + 8));
#pragma unroll
    for (int nt = 0; nt < 4; ++nt) {
      const _Float16* bp = wconv + (nt * 16 + (lane & 15)) * 64 + kc * 32 + khalf * 16;
      acc[nt] = wmma16(a, *(const v16h*)bp, acc[nt]);
    }
  }
  const float* aB = sc + 384; const float* bB = sc + 448;
#pragma unroll
  for (int nt = 0; nt < 4; ++nt) {
    int d = nt * 16 + (lane & 15);
    float av = aB[d], bv = bB[d];
#pragma unroll
    for (int v = 0; v < 8; ++v) {
      int wv = wave * 16 + khalf * 8 + v;
      float val = av * acc[nt][v] + bv;
      tile[wv * 66 + d] = val > 0.f ? val : 0.f;
    }
  }
  __syncthreads();
  int d = tid >> 1, half = tid & 1;
  float* outp = kbuf + ((size_t)(b * 64 + d) * 10 + t) * 4096 + h * 64 + half * 32;
#pragma unroll
  for (int j = 0; j < 32; ++j) outp[j] = tile[(half * 32 + j) * 66 + d];
}

// ---------- MHSA (T=10, hd=8): one thread per (token, head, t) ----------
__global__ void attn_k(const float* qbuf, const float* kbuf, const float* vbuf, _Float16* obuf) {
  int gid = blockIdx.x * blockDim.x + threadIdx.x;
  int n = gid / 80, rm = gid % 80, head = rm / 10, t = rm % 10;
  if (n >= NTOK) return;
  int b = n >> 12, hw = n & 4095;
  const float* qp = qbuf + ((size_t)(b * 10 + t) * 4096 + hw) * 64 + head * 8;
  float q[8];
#pragma unroll
  for (int c = 0; c < 8; ++c) q[c] = qp[c];
  float lg[10], mx = -1e30f;
#pragma unroll
  for (int s = 0; s < 10; ++s) {
    const float* kp = kbuf + (size_t)n * 640 + s * 64 + head * 8;
    float a = 0.f;
#pragma unroll
    for (int c = 0; c < 8; ++c) a += q[c] * kp[c];
    a *= 0.35355339059327373f;          // 1/sqrt(8)
    lg[s] = a; mx = a > mx ? a : mx;
  }
  float sum = 0.f;
#pragma unroll
  for (int s = 0; s < 10; ++s) { lg[s] = __expf(lg[s] - mx); sum += lg[s]; }
  float inv = 1.f / sum;
  float o[8] = {};
#pragma unroll
  for (int s = 0; s < 10; ++s) {
    float p = lg[s] * inv;
    const float* vp = vbuf + (size_t)n * 640 + s * 64 + head * 8;
#pragma unroll
    for (int c = 0; c < 8; ++c) o[c] += p * vp[c];
  }
  _Float16* op = obuf + ((size_t)n * 10 + t) * 64 + head * 8;
#pragma unroll
  for (int c = 0; c < 8; ++c) op[c] = (_Float16)o[c];
}

// ---------- proj 1x1 + bn + relu + residual(x5) -> xresh f16 [NT][64]; M=32/wave ----------
__global__ void __launch_bounds__(128) proj_k(const _Float16* obuf, const _Float16* wproj,
                                              const float* sc, const float* x, _Float16* xresh) {
  int tid = threadIdx.x, lane = tid & 31, wave = tid >> 5, khalf = lane >> 4;
  int r0 = (blockIdx.x * 4 + wave) * 32;
  v8f acc[2][4] = {};
#pragma unroll
  for (int kc = 0; kc < 2; ++kc) {
    const _Float16* ap = obuf + ((size_t)(r0 + (lane & 15))) * 64 + kc * 32 + khalf * 16;
    v16h a0 = mk_afrag(*(const half8*)ap, *(const half8*)(ap + 8));
    v16h a1 = mk_afrag(*(const half8*)(ap + 1024), *(const half8*)(ap + 1032));
#pragma unroll
    for (int nt = 0; nt < 4; ++nt) {
      const _Float16* bp = wproj + (nt * 16 + (lane & 15)) * 64 + kc * 32 + khalf * 16;
      v16h bf = *(const v16h*)bp;
      acc[0][nt] = wmma16(a0, bf, acc[0][nt]);
      acc[1][nt] = wmma16(a1, bf, acc[1][nt]);
    }
  }
  const float* aP = sc + 512; const float* bP = sc + 576;
#pragma unroll
  for (int f = 0; f < 2; ++f) {
#pragma unroll
    for (int nt = 0; nt < 4; ++nt) {
      int d = nt * 16 + (lane & 15);
      float av = aP[d], bv = bP[d];
#pragma unroll
      for (int v = 0; v < 8; ++v) {
        int row = r0 + f * 16 + khalf * 8 + v;
        int n = row / 10, t = row % 10;
        float z = av * acc[f][nt][v] + bv;
        z = z > 0.f ? z : 0.f;
        float res = x[((size_t)n * 64 + d) * 10 + t];
        xresh[(size_t)row * 64 + d] = (_Float16)(res + z);
      }
    }
  }
}

// ---------- ffn1: [NT,64] -> [NT,128], bn + relu; wave = 32 rows x 64 cols ----------
__global__ void __launch_bounds__(128) ffn1_k(const _Float16* xresh, const _Float16* wf1,
                                              const float* sc, _Float16* hbuf) {
  int tid = threadIdx.x, lane = tid & 31, wave = tid >> 5, khalf = lane >> 4;
  int chalf = (wave & 1) * 64;                 // output-channel half
  int r0 = (blockIdx.x * 2 + (wave >> 1)) * 32;
  v8f acc[2][4] = {};
#pragma unroll
  for (int kc = 0; kc < 2; ++kc) {
    const _Float16* ap = xresh + ((size_t)(r0 + (lane & 15))) * 64 + kc * 32 + khalf * 16;
    v16h a0 = mk_afrag(*(const half8*)ap, *(const half8*)(ap + 8));
    v16h a1 = mk_afrag(*(const half8*)(ap + 1024), *(const half8*)(ap + 1032));
#pragma unroll
    for (int nt = 0; nt < 4; ++nt) {
      const _Float16* bp = wf1 + (chalf + nt * 16 + (lane & 15)) * 64 + kc * 32 + khalf * 16;
      v16h bf = *(const v16h*)bp;
      acc[0][nt] = wmma16(a0, bf, acc[0][nt]);
      acc[1][nt] = wmma16(a1, bf, acc[1][nt]);
    }
  }
  const float* a1s = sc + 640; const float* b1s = sc + 768;
#pragma unroll
  for (int f = 0; f < 2; ++f) {
#pragma unroll
    for (int nt = 0; nt < 4; ++nt) {
      int c = chalf + nt * 16 + (lane & 15);
      float av = a1s[c], bv = b1s[c];
#pragma unroll
      for (int v = 0; v < 8; ++v) {
        int row = r0 + f * 16 + khalf * 8 + v;
        float z = av * acc[f][nt][v] + bv;
        hbuf[(size_t)row * 128 + c] = (_Float16)(z > 0.f ? z : 0.f);
      }
    }
  }
}

// ---------- ffn2: [NT,128] -> [NT,64], bn + residual, scatter to [B,D,T,H,W]; M=32/wave ----------
__global__ void __launch_bounds__(128) ffn2_k(const _Float16* hbuf, const _Float16* wf2,
                                              const float* sc, const _Float16* xresh, float* out) {
  int tid = threadIdx.x, lane = tid & 31, wave = tid >> 5, khalf = lane >> 4;
  int r0 = (blockIdx.x * 4 + wave) * 32;
  v8f acc[2][4] = {};
#pragma unroll
  for (int kc = 0; kc < 4; ++kc) {
    const _Float16* ap = hbuf + ((size_t)(r0 + (lane & 15))) * 128 + kc * 32 + khalf * 16;
    v16h a0 = mk_afrag(*(const half8*)ap, *(const half8*)(ap + 8));
    v16h a1 = mk_afrag(*(const half8*)(ap + 2048), *(const half8*)(ap + 2056));
#pragma unroll
    for (int nt = 0; nt < 4; ++nt) {
      const _Float16* bp = wf2 + (nt * 16 + (lane & 15)) * 128 + kc * 32 + khalf * 16;
      v16h bf = *(const v16h*)bp;
      acc[0][nt] = wmma16(a0, bf, acc[0][nt]);
      acc[1][nt] = wmma16(a1, bf, acc[1][nt]);
    }
  }
  const float* a2 = sc + 896; const float* b2 = sc + 960;
#pragma unroll
  for (int f = 0; f < 2; ++f) {
#pragma unroll
    for (int nt = 0; nt < 4; ++nt) {
      int d = nt * 16 + (lane & 15);
      float av = a2[d], bv = b2[d];
#pragma unroll
      for (int v = 0; v < 8; ++v) {
        int row = r0 + f * 16 + khalf * 8 + v;
        int n = row / 10, t = row % 10;
        int b = n >> 12, h = (n >> 6) & 63, w = n & 63;
        float val = (float)xresh[(size_t)row * 64 + d] + av * acc[f][nt][v] + bv;
        out[((size_t)(b * 64 + d) * 10 + t) * 4096 + h * 64 + w] = val;
      }
    }
  }
}

extern "C" void kernel_launch(void* const* d_in, const int* in_sizes, int n_in,
                              void* d_out, int out_size, void* d_ws, size_t ws_size,
                              hipStream_t stream) {
  (void)in_sizes; (void)n_in; (void)out_size; (void)ws_size;
  const float* x      = (const float*)d_in[0];
  const float* WT     = (const float*)d_in[1];
  const float* conv_w = (const float*)d_in[6];
  const float* spdc_w = (const float*)d_in[12];
  const float* temp_w = (const float*)d_in[18];
  const float* proj_w = (const float*)d_in[24];
  const float* f1_w   = (const float*)d_in[30];
  const float* f2_w   = (const float*)d_in[36];

  char* ws = (char*)d_ws;
  _Float16* xhwc  = (_Float16*)(ws);                 // later reused as obuf
  _Float16* obuf  = xhwc;
  _Float16* q1h   = (_Float16*)(ws + 2 * SZ);        // later reused as xresh
  _Float16* xresh = q1h;
  float*    qbuf  = (float*)(ws + 4 * SZ);
  float*    kbuf  = (float*)(ws + 8 * SZ);
  float*    vbuf  = (float*)(ws + 12 * SZ);
  _Float16* hbuf  = (_Float16*)(ws + 12 * SZ);       // reuses vbuf after attention
  _Float16* ybuf  = (_Float16*)(ws + 16 * SZ);
  _Float16* wb    = (_Float16*)(ws + 18 * SZ);
  const _Float16* wspdc = wb;
  const _Float16* wtemp = wb + 102400;
  const _Float16* wconv = wb + 122880;
  const _Float16* wproj = wb + 126976;
  const _Float16* wf1   = wb + 131072;
  const _Float16* wf2   = wb + 139264;
  float* sc = (float*)(ws + 18 * SZ + 2ull * 147456);

  ScaleArgs A;
  const int gi[7]   = {14, 20, 2, 8, 26, 32, 38};
  const int bi[7]   = {13, 19, -1, 7, 25, 31, 37};
  const int cnt7[7] = {64, 64, 64, 64, 64, 128, 64};
  const int off7[7] = {0, 128, 256, 384, 512, 640, 896};
  for (int s = 0; s < 7; ++s) {
    A.g[s]  = (const float*)d_in[gi[s]];
    A.be[s] = (const float*)d_in[gi[s] + 1];
    A.mn[s] = (const float*)d_in[gi[s] + 2];
    A.vr[s] = (const float*)d_in[gi[s] + 3];
    A.bias[s] = bi[s] >= 0 ? (const float*)d_in[bi[s]] : nullptr;
    A.cnt[s] = cnt7[s];
    A.off[s] = off7[s];
  }
  A.out = sc;

  prep_scales_k<<<2, 256, 0, stream>>>(A);
  prep_weights_k<<<576, 256, 0, stream>>>(spdc_w, temp_w, conv_w, proj_w, f1_w, f2_w, wb);
  prep_xv_k<<<4096, 256, 0, stream>>>(x, xhwc, vbuf);
  spdc_k<<<1280, 128, 0, stream>>>(xhwc, wspdc, sc, q1h);
  temp_k<<<2560, 128, 0, stream>>>(q1h, wtemp, sc, qbuf);
  wt1_k<<<4096, 256, 0, stream>>>(x, WT, sc, ybuf);
  wt2_k<<<2560, 128, 0, stream>>>(ybuf, wconv, sc, kbuf);
  attn_k<<<5120, 256, 0, stream>>>(qbuf, kbuf, vbuf, obuf);
  proj_k<<<1280, 128, 0, stream>>>(obuf, wproj, sc, x, xresh);
  ffn1_k<<<2560, 128, 0, stream>>>(xresh, wf1, sc, hbuf);
  ffn2_k<<<1280, 128, 0, stream>>>(hbuf, wf2, sc, xresh, (float*)d_out);
}
